// CVRPMoveScorer_58093727646192
// MI455X (gfx1250) — compile-verified, matrix operand order checked by reading, not run on hardware
//
#include <hip/hip_runtime.h>
#include <math.h>

// ---------------------------------------------------------------------------
// CVRP move scorer (GNN) for gfx1250: bf16 WMMA (16x16x32, fp32 accum) for all
// GEMMs; fp32 gathers / atomics / LayerNorm / SiLU around them. wave32.
// SiLU/LN use v_rcp_f32 / v_rsq_f32 hardware transcendentals (co-issue with
// the matrix pipe) instead of IEEE div/sqrt expansion sequences.
// ---------------------------------------------------------------------------

typedef __bf16 bf16t;
typedef __attribute__((ext_vector_type(16))) __bf16 v16bf;
typedef __attribute__((ext_vector_type(8)))  float  v8f;

#define NB_   64      // batch
#define NN_   1024    // nodes
#define NE_   2048    // edges
#define NM_   4096    // moves
#define NF_   7       // input features
#define DD_   64      // hidden dim
#define LSTR  68      // LDS tile row stride (floats): 17*4 -> conflict-free

__device__ __forceinline__ float silu_f(float x) {
  // x * sigmoid(x); v_exp_f32 + v_rcp_f32 (TRANS ops, overlap with WMMA)
  return x * __builtin_amdgcn_rcpf(1.0f + __expf(-x));
}

// --- B fragment: packed bf16 weights, 512 halfs per 32x16 tile, lane-major ---
__device__ __forceinline__ v16bf ldB(const bf16t* wp, int kt, int nt, int lane) {
  return *(const v16bf*)(wp + ((((kt << 2) + nt) << 5) + lane) * 16);
}

// --- A fragment (16x32) from a per-lane fp32 row pointer (row = lane&15) ---
// lane<16: K = k0+0..7 , k0+16..23 ; lane>=16: K = k0+8..15 , k0+24..31
__device__ __forceinline__ v16bf ldA_g(const float* row, int k0, int lane) {
  const int kb = k0 + ((lane >= 16) ? 8 : 0);
  const float4 x0 = *(const float4*)(row + kb);
  const float4 x1 = *(const float4*)(row + kb + 4);
  const float4 y0 = *(const float4*)(row + kb + 16);
  const float4 y1 = *(const float4*)(row + kb + 20);
  v16bf a;
  a[0]=(bf16t)x0.x;  a[1]=(bf16t)x0.y;  a[2]=(bf16t)x0.z;  a[3]=(bf16t)x0.w;
  a[4]=(bf16t)x1.x;  a[5]=(bf16t)x1.y;  a[6]=(bf16t)x1.z;  a[7]=(bf16t)x1.w;
  a[8]=(bf16t)y0.x;  a[9]=(bf16t)y0.y;  a[10]=(bf16t)y0.z; a[11]=(bf16t)y0.w;
  a[12]=(bf16t)y1.x; a[13]=(bf16t)y1.y; a[14]=(bf16t)y1.z; a[15]=(bf16t)y1.w;
  return a;
}

// --- A fragment from the wave's LDS tile (16 x 64, stride LSTR) ---
__device__ __forceinline__ v16bf ldA_s(const float* s, int k0, int lane) {
  const int r  = lane & 15;
  const int kb = k0 + ((lane >= 16) ? 8 : 0);
  const float* row = s + r * LSTR;
  v16bf a;
#pragma unroll
  for (int j = 0; j < 8; ++j) {
    a[j]     = (bf16t)row[kb + j];
    a[8 + j] = (bf16t)row[kb + 16 + j];
  }
  return a;
}

// --- store 16x64 accumulator (+bias, optional SiLU) into the LDS tile ---
__device__ __forceinline__ void stC_s(float* s, const v8f* acc, const float* bias,
                                      bool act, int lane) {
  const int c0 = lane & 15;
  const int r0 = (lane >= 16) ? 8 : 0;
#pragma unroll
  for (int nt = 0; nt < 4; ++nt) {
    const int col = nt * 16 + c0;
    const float bb = bias[col];
#pragma unroll
    for (int r = 0; r < 8; ++r) {
      float v = acc[nt][r] + bb;
      if (act) v = silu_f(v);
      s[(r0 + r) * LSTR + col] = v;
    }
  }
}

// ---------------------------------------------------------------------------
// weight pre-pack: fp32 (K x 64) -> bf16 B-fragment layout (Kpad/32 x 4 tiles)
// ---------------------------------------------------------------------------
__global__ void k_pack(const float* __restrict__ w, bf16t* __restrict__ dst,
                       int Ksrc, int Kpad) {
  const int total = Kpad * 64;
  for (int e = blockIdx.x * blockDim.x + threadIdx.x; e < total;
       e += gridDim.x * blockDim.x) {
    const int f = e >> 9, rr = e & 511;
    const int lane = rr >> 4, j = rr & 15;
    const int kt = f >> 2, nt = f & 3;
    const int k = kt * 32 + ((lane >= 16) ? 8 : 0) + ((j < 8) ? j : 16 + (j - 8));
    const int n = nt * 16 + (lane & 15);
    dst[e] = (bf16t)((k < Ksrc) ? w[k * 64 + n] : 0.0f);
  }
}

__global__ void k_zero(float* __restrict__ p, int n) {
  for (int i = blockIdx.x * blockDim.x + threadIdx.x; i < n;
       i += gridDim.x * blockDim.x) p[i] = 0.0f;
}

// ---------------------------------------------------------------------------
// embed: h = W1 * silu(W0 * x + b0) + b1     (7 -> 64 -> 64)
// ---------------------------------------------------------------------------
__global__ __launch_bounds__(256) void k_embed(
    const float* __restrict__ nf, const bf16t* __restrict__ w0p,
    const float* __restrict__ b0, const bf16t* __restrict__ w1p,
    const float* __restrict__ b1, float* __restrict__ h) {
  __shared__ float smem[8 * 16 * LSTR];
  const int lane = threadIdx.x & 31, wv = threadIdx.x >> 5;
  float* s = smem + wv * 16 * LSTR;
  const long tile = (long)blockIdx.x * 8 + wv;     // 16-row tile over B*N1
  const long row0 = tile * 16;
  const int r = lane & 15;
  const float* xrow = nf + (row0 + r) * NF_;

  v8f acc[4] = {};
  {
    const int kb = (lane >= 16) ? 8 : 0;
    v16bf a;
#pragma unroll
    for (int j = 0; j < 8; ++j) {
      const int k1 = kb + j, k2 = kb + 16 + j;
      a[j]     = (bf16t)((k1 < NF_) ? xrow[k1] : 0.0f);
      a[8 + j] = (bf16t)((k2 < NF_) ? xrow[k2] : 0.0f);
    }
#pragma unroll
    for (int nt = 0; nt < 4; ++nt)
      acc[nt] = __builtin_amdgcn_wmma_f32_16x16x32_bf16(
          false, a, false, ldB(w0p, 0, nt, lane), (short)0, acc[nt], false, false);
  }
  stC_s(s, acc, b0, true, lane);
  __syncthreads();

  v8f acc2[4] = {};
#pragma unroll
  for (int c = 0; c < 2; ++c) {
    const v16bf a = ldA_s(s, c * 32, lane);
#pragma unroll
    for (int nt = 0; nt < 4; ++nt)
      acc2[nt] = __builtin_amdgcn_wmma_f32_16x16x32_bf16(
          false, a, false, ldB(w1p, c, nt, lane), (short)0, acc2[nt], false, false);
  }
  const int c0 = lane & 15, r0 = (lane >= 16) ? 8 : 0;
#pragma unroll
  for (int nt = 0; nt < 4; ++nt) {
    const int col = nt * 16 + c0;
    const float bb = b1[col];
#pragma unroll
    for (int rr = 0; rr < 8; ++rr)
      h[(row0 + r0 + rr) * DD_ + col] = acc2[nt][rr] + bb;
  }
}

// ---------------------------------------------------------------------------
// msg: per-edge MLP on [h_tgt, h_src] (128->64->64) + atomic scatter into agg
// ---------------------------------------------------------------------------
__global__ __launch_bounds__(256) void k_msg(
    const float* __restrict__ h, const int* __restrict__ eidx,
    const bf16t* __restrict__ w0p, const float* __restrict__ b0,
    const bf16t* __restrict__ w1p, const float* __restrict__ b1,
    float* __restrict__ agg) {
  __shared__ float smem[8 * 16 * LSTR];
  __shared__ int tgts[8][16];
  const int lane = threadIdx.x & 31, wv = threadIdx.x >> 5;
  float* s = smem + wv * 16 * LSTR;
  const long tile = (long)blockIdx.x * 8 + wv;     // 16-edge tile over B*E
  const long b = tile / (NE_ / 16);
  const long e0 = (tile % (NE_ / 16)) * 16;
  const int r = lane & 15;
  const int* ep = eidx + ((b * NE_ + e0 + r) << 1);
  const int srcN = ep[0], tgtN = ep[1];
  const float* hb   = h + b * (long)NN_ * DD_;
  const float* rowT = hb + (long)tgtN * DD_;
  const float* rowS = hb + (long)srcN * DD_;
  if (lane < 16) tgts[wv][lane] = tgtN;

  v8f acc[4] = {};
#pragma unroll
  for (int c = 0; c < 4; ++c) {
    const v16bf a = ldA_g((c < 2) ? rowT : rowS, (c & 1) * 32, lane);
#pragma unroll
    for (int nt = 0; nt < 4; ++nt)
      acc[nt] = __builtin_amdgcn_wmma_f32_16x16x32_bf16(
          false, a, false, ldB(w0p, c, nt, lane), (short)0, acc[nt], false, false);
  }
  stC_s(s, acc, b0, true, lane);
  __syncthreads();

  v8f acc2[4] = {};
#pragma unroll
  for (int c = 0; c < 2; ++c) {
    const v16bf a = ldA_s(s, c * 32, lane);
#pragma unroll
    for (int nt = 0; nt < 4; ++nt)
      acc2[nt] = __builtin_amdgcn_wmma_f32_16x16x32_bf16(
          false, a, false, ldB(w1p, c, nt, lane), (short)0, acc2[nt], false, false);
  }
  const int c0 = lane & 15, r0 = (lane >= 16) ? 8 : 0;
  float* aggb = agg + b * (long)NN_ * DD_;
#pragma unroll
  for (int nt = 0; nt < 4; ++nt) {
    const int col = nt * 16 + c0;
    const float bb = b1[col];
#pragma unroll
    for (int rr = 0; rr < 8; ++rr)
      atomicAdd(aggb + (long)tgts[wv][r0 + rr] * DD_ + col, acc2[nt][rr] + bb);
  }
}

// ---------------------------------------------------------------------------
// upd: h = LayerNorm(h + MLP([h, agg])) in place   (128->64->64)
// ---------------------------------------------------------------------------
__global__ __launch_bounds__(256) void k_upd(
    float* __restrict__ h, const float* __restrict__ agg,
    const bf16t* __restrict__ w0p, const float* __restrict__ b0,
    const bf16t* __restrict__ w1p, const float* __restrict__ b1,
    const float* __restrict__ lng, const float* __restrict__ lnb) {
  __shared__ float smem[8 * 16 * LSTR];
  const int lane = threadIdx.x & 31, wv = threadIdx.x >> 5;
  float* s = smem + wv * 16 * LSTR;
  const long tile = (long)blockIdx.x * 8 + wv;     // 16-node tile over B*N1
  const long row0 = tile * 16;
  const int r = lane & 15;
  const float* rowH = h   + (row0 + r) * DD_;
  const float* rowA = agg + (row0 + r) * DD_;

  v8f acc[4] = {};
#pragma unroll
  for (int c = 0; c < 4; ++c) {
    const v16bf a = ldA_g((c < 2) ? rowH : rowA, (c & 1) * 32, lane);
#pragma unroll
    for (int nt = 0; nt < 4; ++nt)
      acc[nt] = __builtin_amdgcn_wmma_f32_16x16x32_bf16(
          false, a, false, ldB(w0p, c, nt, lane), (short)0, acc[nt], false, false);
  }
  stC_s(s, acc, b0, true, lane);
  __syncthreads();

  v8f acc2[4] = {};
#pragma unroll
  for (int c = 0; c < 2; ++c) {
    const v16bf a = ldA_s(s, c * 32, lane);
#pragma unroll
    for (int nt = 0; nt < 4; ++nt)
      acc2[nt] = __builtin_amdgcn_wmma_f32_16x16x32_bf16(
          false, a, false, ldB(w1p, c, nt, lane), (short)0, acc2[nt], false, false);
  }
  __syncthreads();                                   // x1 reads done
  const int c0 = lane & 15, r0 = (lane >= 16) ? 8 : 0;
#pragma unroll
  for (int nt = 0; nt < 4; ++nt) {                   // residual into LDS
    const int col = nt * 16 + c0;
    const float bb = b1[col];
#pragma unroll
    for (int rr = 0; rr < 8; ++rr)
      s[(r0 + rr) * LSTR + col] = acc2[nt][rr] + bb + h[(row0 + r0 + rr) * DD_ + col];
  }
  __syncthreads();

  if (lane < 16) {                                   // LayerNorm, one row/lane
    const float* rowv = s + lane * LSTR;
    float m = 0.0f;
#pragma unroll
    for (int j = 0; j < DD_; ++j) m += rowv[j];
    m *= (1.0f / DD_);
    float v = 0.0f;
#pragma unroll
    for (int j = 0; j < DD_; ++j) { const float d = rowv[j] - m; v += d * d; }
    const float rstd = __builtin_amdgcn_rsqf(v * (1.0f / DD_) + 1e-5f); // v_rsq_f32
    float* outp = h + (row0 + lane) * DD_;
#pragma unroll
    for (int j = 0; j < DD_; j += 4) {
      float4 o;
      o.x = (rowv[j + 0] - m) * rstd * lng[j + 0] + lnb[j + 0];
      o.y = (rowv[j + 1] - m) * rstd * lng[j + 1] + lnb[j + 1];
      o.z = (rowv[j + 2] - m) * rstd * lng[j + 2] + lnb[j + 2];
      o.w = (rowv[j + 3] - m) * rstd * lng[j + 3] + lnb[j + 3];
      *(float4*)(outp + j) = o;
    }
  }
}

// ---------------------------------------------------------------------------
// scorer: gather 4 node rows (256) -> 64 (silu) -> 64 (silu) -> 1, mask
// ---------------------------------------------------------------------------
__global__ __launch_bounds__(256) void k_score(
    const float* __restrict__ h, const int* __restrict__ mv,
    const unsigned char* __restrict__ mask,
    const bf16t* __restrict__ w0p, const float* __restrict__ b0,
    const bf16t* __restrict__ w1p, const float* __restrict__ b1,
    const float* __restrict__ w2, const float* __restrict__ b2,
    float* __restrict__ out) {
  __shared__ float smem[8 * 16 * LSTR];
  const int lane = threadIdx.x & 31, wv = threadIdx.x >> 5;
  float* s = smem + wv * 16 * LSTR;
  const long tile = (long)blockIdx.x * 8 + wv;     // 16-move tile over B*M
  const long b = tile / (NM_ / 16);
  const long m0 = (tile % (NM_ / 16)) * 16;
  const int r = lane & 15;
  const int4 nd = *(const int4*)(mv + ((b * NM_ + m0 + r) << 2));
  const float* hb = h + b * (long)NN_ * DD_;
  const float* rows[4] = { hb + (long)nd.x * DD_, hb + (long)nd.y * DD_,
                           hb + (long)nd.z * DD_, hb + (long)nd.w * DD_ };

  v8f acc[4] = {};
#pragma unroll
  for (int c = 0; c < 8; ++c) {
    const v16bf a = ldA_g(rows[c >> 1], (c & 1) * 32, lane);
#pragma unroll
    for (int nt = 0; nt < 4; ++nt)
      acc[nt] = __builtin_amdgcn_wmma_f32_16x16x32_bf16(
          false, a, false, ldB(w0p, c, nt, lane), (short)0, acc[nt], false, false);
  }
  stC_s(s, acc, b0, true, lane);
  __syncthreads();

  v8f acc2[4] = {};
#pragma unroll
  for (int c = 0; c < 2; ++c) {
    const v16bf a = ldA_s(s, c * 32, lane);
#pragma unroll
    for (int nt = 0; nt < 4; ++nt)
      acc2[nt] = __builtin_amdgcn_wmma_f32_16x16x32_bf16(
          false, a, false, ldB(w1p, c, nt, lane), (short)0, acc2[nt], false, false);
  }
  __syncthreads();                                   // x1 reads done
  stC_s(s, acc2, b1, true, lane);                    // x2 = silu(.) into LDS
  __syncthreads();

  if (lane < 16) {                                   // final 64-dot, one row/lane
    const float* rowv = s + lane * LSTR;
    float d = 0.0f;
#pragma unroll
    for (int j = 0; j < DD_; ++j) d += rowv[j] * w2[j];
    d += b2[0];
    const long mi = b * NM_ + m0 + lane;
    out[mi] = mask[mi] ? d : -INFINITY;
  }
}

// ---------------------------------------------------------------------------
extern "C" void kernel_launch(void* const* d_in, const int* in_sizes, int n_in,
                              void* d_out, int out_size, void* d_ws, size_t ws_size,
                              hipStream_t stream) {
  (void)in_sizes; (void)n_in; (void)out_size; (void)ws_size;
  const float* nf   = (const float*)d_in[0];
  const int*   eidx = (const int*)d_in[1];
  const int*   mv   = (const int*)d_in[2];
  const unsigned char* mask = (const unsigned char*)d_in[3];
  const float* e0b = (const float*)d_in[4];
  const float* e0w = (const float*)d_in[5];
  const float* e1b = (const float*)d_in[6];
  const float* e1w = (const float*)d_in[7];
  const float* s0b = (const float*)d_in[48];
  const float* s0w = (const float*)d_in[49];
  const float* s1b = (const float*)d_in[50];
  const float* s1w = (const float*)d_in[51];
  const float* s2b = (const float*)d_in[52];
  const float* s2w = (const float*)d_in[53];

  // workspace: h (16MB) | agg (16MB) | packed bf16 weights (~250KB)
  float* h   = (float*)d_ws;
  float* agg = h + (size_t)NB_ * NN_ * DD_;
  bf16t* wpk = (bf16t*)(agg + (size_t)NB_ * NN_ * DD_);
  bf16t* pe0 = wpk;                 // 32x64
  bf16t* pe1 = pe0 + 2048;          // 64x64
  bf16t* pl  = pe1 + 4096;          // 4 layers x (8192+4096+8192+4096)
  bf16t* ps0 = pl + 4 * 24576;      // 256x64
  bf16t* ps1 = ps0 + 16384;         // 64x64

  k_pack<<<8, 256, 0, stream>>>(e0w, pe0, NF_, 32);
  k_pack<<<16, 256, 0, stream>>>(e1w, pe1, 64, 64);
  for (int l = 0; l < 4; ++l) {
    const int base = 8 + 10 * l;
    bf16t* lm0 = pl + l * 24576;
    k_pack<<<32, 256, 0, stream>>>((const float*)d_in[base + 3], lm0,          128, 128);
    k_pack<<<16, 256, 0, stream>>>((const float*)d_in[base + 5], lm0 + 8192,    64,  64);
    k_pack<<<32, 256, 0, stream>>>((const float*)d_in[base + 7], lm0 + 12288,  128, 128);
    k_pack<<<16, 256, 0, stream>>>((const float*)d_in[base + 9], lm0 + 20480,   64,  64);
  }
  k_pack<<<64, 256, 0, stream>>>(s0w, ps0, 256, 256);
  k_pack<<<16, 256, 0, stream>>>(s1w, ps1, 64, 64);

  // embed: B*N1/16 = 4096 wave tiles, 8 waves/block
  k_embed<<<512, 256, 0, stream>>>(nf, pe0, e0b, pe1, e1b, h);

  for (int l = 0; l < 4; ++l) {
    const int base = 8 + 10 * l;
    const float* lnb_ = (const float*)d_in[base + 0];
    const float* lng_ = (const float*)d_in[base + 1];
    const float* m0b  = (const float*)d_in[base + 2];
    const float* m1b  = (const float*)d_in[base + 4];
    const float* u0b  = (const float*)d_in[base + 6];
    const float* u1b  = (const float*)d_in[base + 8];
    bf16t* lm0 = pl + l * 24576;
    bf16t* lm1 = lm0 + 8192;
    bf16t* lu0 = lm0 + 12288;
    bf16t* lu1 = lm0 + 20480;

    k_zero<<<4096, 256, 0, stream>>>(agg, NB_ * NN_ * DD_);
    // B*E/16 = 8192 wave tiles
    k_msg<<<1024, 256, 0, stream>>>(h, eidx, lm0, m0b, lm1, m1b, agg);
    // B*N1/16 = 4096 wave tiles
    k_upd<<<512, 256, 0, stream>>>(h, agg, lu0, u0b, lu1, u1b, lng_, lnb_);
  }

  // B*M/16 = 16384 wave tiles
  k_score<<<2048, 256, 0, stream>>>(h, mv, mask, ps0, s0b, ps1, s1b,
                                    s2w, s2b, (float*)d_out);
}
